// ContextUNETR_41540923687612
// MI455X (gfx1250) — compile-verified, matrix operand order checked by reading, not run on hardware
//
#include <hip/hip_runtime.h>

#define LNUM 5
#define T_   2048
#define D_   512
#define E_   8
#define K_   2
#define H_   2048
#define LT_  (LNUM * T_)     // 10240 tokens
#define TM   32              // tokens per GEMM tile
#define HC   64              // hidden (H) chunk per fused step
#define NCH  (H_ / HC)       // 32 chunks
#define NBLK_GEMM 640

typedef __attribute__((ext_vector_type(16))) __bf16 v16bf;
typedef __attribute__((ext_vector_type(8)))  __bf16 v8bf;
typedef __attribute__((ext_vector_type(8)))  float  v8f;
typedef __attribute__((ext_vector_type(4)))  unsigned int v4u;
typedef __attribute__((ext_vector_type(8)))  int v8i;
typedef __attribute__((ext_vector_type(4)))  int v4i;

// Device pass MUST have the TDM builtins (unambiguous signal in the binary).
// Host pass only needs a stub (it never codegens device builtins).
#if defined(__HIP_DEVICE_COMPILE__)
# if __has_builtin(__builtin_amdgcn_tensor_load_to_lds) && __has_builtin(__builtin_amdgcn_s_wait_tensorcnt)
#  define HAVE_TDM 1
# else
#  error "gfx1250 TDM builtins (__builtin_amdgcn_tensor_load_to_lds / s_wait_tensorcnt) not available"
# endif
#else
# define HAVE_TDM 0
#endif

// ---------------- workspace layout (bytes) ----------------
static constexpr size_t XBF_OFF  = 0;
static constexpr size_t XBF_SZ   = (size_t)LT_ * D_ * 2;          // x in bf16
static constexpr size_t W1T_OFF  = XBF_OFF + XBF_SZ;
static constexpr size_t W1T_SZ   = (size_t)E_ * H_ * D_ * 2;      // W1^T [E][H][D] bf16
static constexpr size_t W2T_OFF  = W1T_OFF + W1T_SZ;
static constexpr size_t W2T_SZ   = (size_t)E_ * D_ * H_ * 2;      // W2^T [E][D][H] bf16
static constexpr size_t TOKE_OFF = W2T_OFF + W2T_SZ;              // int  [LT][K]
static constexpr size_t TOKG_OFF = TOKE_OFF + (size_t)LT_ * K_ * 4; // f32 [LT][K]
static constexpr size_t PERM_OFF = TOKG_OFF + (size_t)LT_ * K_ * 4; // int  [LT*K]
static constexpr size_t PGAT_OFF = PERM_OFF + (size_t)LT_ * K_ * 4; // f32 [LT*K]
static constexpr size_t META_OFF = PGAT_OFF + (size_t)LT_ * K_ * 4;
// meta ints: counts[8] cursor[8] offsets[9] tilesPrefix[9] totalTiles[1]

// ---------------- device helpers ----------------
__device__ __forceinline__ __bf16 f2bf(float f) {
  unsigned u = __builtin_bit_cast(unsigned, f);
  unsigned r = (u + 0x7FFFu + ((u >> 16) & 1u)) >> 16;
  return __builtin_bit_cast(__bf16, (unsigned short)r);
}

__device__ __forceinline__ v16bf cat8(v8bf lo, v8bf hi) {
  return __builtin_shufflevector(lo, hi, 0,1,2,3,4,5,6,7,8,9,10,11,12,13,14,15);
}

// Load one 16x32 bf16 WMMA operand (A or B^T) from a row-major LDS tile.
// lane<16  : row = base+(lane&15), K = {k0..k0+7} U {k0+16..k0+23}
// lane>=16 : same rows, K offset +8 (per CDNA5 ISA 16-bit operand layout)
__device__ __forceinline__ v16bf ld_op(const __bf16* p) {
  v8bf lo = *(const v8bf*)p;
  v8bf hi = *(const v8bf*)(p + 16);
  return cat8(lo, hi);
}

__device__ __forceinline__ unsigned rfl(unsigned v) {
  return (unsigned)__builtin_amdgcn_readfirstlane((int)v);
}

// 2-D TDM tile load: tile0 contiguous bf16 elements per row, tile1 rows,
// row stride stride0 (elements). Descriptor per CDNA5 ISA ch.8 (D# groups).
__device__ __forceinline__ void tdm_load_2d(unsigned lds_addr, const void* gptr,
                                            unsigned tile0, unsigned tile1,
                                            unsigned long long stride0) {
#if HAVE_TDM
  unsigned long long ga = (unsigned long long)gptr;
  unsigned glo = rfl((unsigned)ga);
  unsigned ghi = rfl((unsigned)(ga >> 32));
  unsigned la  = rfl(lds_addr);
  v4u g0;
  g0[0] = 1u;                                   // count=1, user mode, no gather
  g0[1] = la;                                   // lds_addr
  g0[2] = glo;                                  // global_addr[31:0]
  g0[3] = (ghi & 0x1FFFFFFu) | (2u << 30);      // global_addr[56:32] | type=2
  unsigned dim0 = tile0, dim1 = tile1;
  v8i g1;
  g1[0] = (int)(1u << 16);                      // wg_mask=0, data_size=1 (2B)
  g1[1] = (int)((dim0 & 0xFFFFu) << 16);        // tensor_dim0[15:0]
  g1[2] = (int)(((dim0 >> 16) & 0xFFFFu) | ((dim1 & 0xFFFFu) << 16));
  g1[3] = (int)(((dim1 >> 16) & 0xFFFFu) | ((tile0 & 0xFFFFu) << 16)); // tile_dim0
  g1[4] = (int)(tile1 & 0xFFFFu);               // tile_dim1 (tile_dim2=0)
  g1[5] = (int)(unsigned)(stride0 & 0xFFFFFFFFu);       // tensor_dim0_stride lo
  g1[6] = (int)(unsigned)((stride0 >> 32) & 0xFFFFu);   // stride0 hi, stride1=0
  g1[7] = 0;
  v4i z = {0, 0, 0, 0};
#if defined(__clang_major__) && (__clang_major__ >= 23)
  v8i z8 = {0, 0, 0, 0, 0, 0, 0, 0};
  __builtin_amdgcn_tensor_load_to_lds(g0, g1, z, z, z8, 0);
#else
  __builtin_amdgcn_tensor_load_to_lds(g0, g1, z, z, 0);
#endif
#else
  (void)lds_addr; (void)gptr; (void)tile0; (void)tile1; (void)stride0;
#endif
}

__device__ __forceinline__ void tdm_wait(int pending) {
#if HAVE_TDM
  if (pending) __builtin_amdgcn_s_wait_tensorcnt(2);   // keep 2 in flight
  else         __builtin_amdgcn_s_wait_tensorcnt(0);   // drain
#else
  (void)pending;
#endif
}

// ---------------- trivial kernels ----------------
__global__ void zero_f32(float* p, int n) {
  for (int i = blockIdx.x * blockDim.x + threadIdx.x; i < n; i += gridDim.x * blockDim.x)
    p[i] = 0.f;
}
__global__ void zero_i32(int* p, int n) {
  for (int i = blockIdx.x * blockDim.x + threadIdx.x; i < n; i += gridDim.x * blockDim.x)
    p[i] = 0;
}
__global__ void cvt_x(const float* __restrict__ x, __bf16* __restrict__ xb, int n) {
  for (int i = blockIdx.x * blockDim.x + threadIdx.x; i < n; i += gridDim.x * blockDim.x)
    xb[i] = f2bf(x[i]);
}
__global__ void cvt_w1(const float* __restrict__ W1, __bf16* __restrict__ w1t) {
  const int n = E_ * H_ * D_;
  for (int i = blockIdx.x * blockDim.x + threadIdx.x; i < n; i += gridDim.x * blockDim.x) {
    int e = i / (H_ * D_);
    int r = i - e * (H_ * D_);
    int h = r / D_;
    int d = r - h * D_;
    w1t[i] = f2bf(W1[(size_t)e * D_ * H_ + (size_t)d * H_ + h]);   // [E][H][D] <- [E][D][H]
  }
}
__global__ void cvt_w2(const float* __restrict__ W2, __bf16* __restrict__ w2t) {
  const int n = E_ * D_ * H_;
  for (int i = blockIdx.x * blockDim.x + threadIdx.x; i < n; i += gridDim.x * blockDim.x) {
    int e = i / (D_ * H_);
    int r = i - e * (D_ * H_);
    int d = r / H_;
    int h = r - d * H_;
    w2t[i] = f2bf(W2[(size_t)e * H_ * D_ + (size_t)h * D_ + d]);   // [E][D][H] <- [E][H][D]
  }
}

// ---------------- router: noisy top-2 ----------------
__global__ void router_kernel(const float* __restrict__ x, const float* __restrict__ noise,
                              const float* __restrict__ Wr, const float* __restrict__ br,
                              const float* __restrict__ Wn, const float* __restrict__ bn,
                              int* __restrict__ topi_out, int* __restrict__ tokExp,
                              float* __restrict__ tokGate, int* __restrict__ counts) {
  int t = blockIdx.x * blockDim.x + threadIdx.x;
  if (t >= LT_) return;
  int l = t / T_;
  const float* xr = x + (size_t)t * D_;
  const float* wr = Wr + (size_t)l * D_ * E_;
  const float* wn = Wn + (size_t)l * D_ * E_;
  float lg[E_], nl[E_];
#pragma unroll
  for (int e = 0; e < E_; ++e) { lg[e] = br[l * E_ + e]; nl[e] = bn[l * E_ + e]; }
  for (int d = 0; d < D_; ++d) {
    float xv = xr[d];
    const float* wrd = wr + d * E_;
    const float* wnd = wn + d * E_;
#pragma unroll
    for (int e = 0; e < E_; ++e) { lg[e] = fmaf(xv, wrd[e], lg[e]); nl[e] = fmaf(xv, wnd[e], nl[e]); }
  }
  const float* nz = noise + (size_t)t * E_;
  float v[E_];
#pragma unroll
  for (int e = 0; e < E_; ++e) {
    float s = nl[e];
    float sp = (s > 20.f) ? s : log1pf(expf(s));      // softplus
    v[e] = lg[e] + nz[e] * sp;
  }
  int i0 = 0; float v0 = v[0];
#pragma unroll
  for (int e = 1; e < E_; ++e) if (v[e] > v0) { v0 = v[e]; i0 = e; }
  int i1 = -1; float v1 = -3.4e38f;
#pragma unroll
  for (int e = 0; e < E_; ++e) if (e != i0 && v[e] > v1) { v1 = v[e]; i1 = e; }
  float g0 = 1.f / (1.f + expf(v1 - v0));
  float g1 = 1.f - g0;
  topi_out[t * K_ + 0] = i0;  topi_out[t * K_ + 1] = i1;
  tokExp[t * K_ + 0] = i0;    tokExp[t * K_ + 1] = i1;
  tokGate[t * K_ + 0] = g0;   tokGate[t * K_ + 1] = g1;
  atomicAdd(&counts[i0], 1);
  atomicAdd(&counts[i1], 1);
}

__global__ void prefix_kernel(const int* __restrict__ counts, int* __restrict__ offsets,
                              int* __restrict__ tilesPrefix, int* __restrict__ totalTiles) {
  int off = 0, tp = 0;
  for (int e = 0; e < E_; ++e) {
    offsets[e] = off;
    tilesPrefix[e] = tp;
    off += counts[e];
    tp += (counts[e] + TM - 1) / TM;
  }
  offsets[E_] = off;
  tilesPrefix[E_] = tp;
  *totalTiles = tp;
}

__global__ void scatter_kernel(const int* __restrict__ tokExp, const float* __restrict__ tokGate,
                               const int* __restrict__ offsets, int* __restrict__ cursor,
                               int* __restrict__ perm, float* __restrict__ pgate) {
  int t = blockIdx.x * blockDim.x + threadIdx.x;
  if (t >= LT_) return;
#pragma unroll
  for (int k = 0; k < K_; ++k) {
    int e = tokExp[t * K_ + k];
    int pos = atomicAdd(&cursor[e], 1);
    perm[offsets[e] + pos] = t;
    pgate[offsets[e] + pos] = tokGate[t * K_ + k];
  }
}

// ------- grouped fused MoE MLP: bf16 WMMA + double-buffered TDM weights -------
__global__ __launch_bounds__(256) void moe_gemm_kernel(
    const __bf16* __restrict__ xbf, const __bf16* __restrict__ w1t,
    const __bf16* __restrict__ w2t, const float* __restrict__ b1,
    const float* __restrict__ b2, const int* __restrict__ perm,
    const float* __restrict__ pgate, const int* __restrict__ offsets,
    const int* __restrict__ tilesPrefix, const int* __restrict__ totalTilesPtr,
    float* __restrict__ out) {
  __shared__ __bf16 w1s[2][HC][D_];  // 2 x 64 KB  W1^T chunk [h][d]
  __shared__ __bf16 w2s[2][D_][HC];  // 2 x 64 KB  W2^T chunk [d][h]
  __shared__ __bf16 xs[TM][D_];      // 32 KB gathered token rows
  __shared__ __bf16 hs[TM][HC];      //  4 KB relu(hidden) chunk
  __shared__ int   rows_s[TM];
  __shared__ float gates_s[TM];
  // total: 292.25 KB < 320 KB/WGP

  const int tid  = threadIdx.x;
  const int lane = tid & 31;
  const int wave = tid >> 5;
  const int lrow = lane & 15;
  const int ksel = (lane >> 4) * 8;
  const int total = *totalTilesPtr;

  for (int tile = blockIdx.x; tile < total; tile += gridDim.x) {
    int e = 0;
#pragma unroll
    for (int q = 1; q < E_; ++q) if (tile >= tilesPrefix[q]) e = q;
    const int tileM = tile - tilesPrefix[e];
    const int aBase = offsets[e] + tileM * TM;
    const int aEnd  = offsets[e + 1];

    __syncthreads();                       // prev tile fully done with LDS
    if (tid < TM) {
      int ai = aBase + tid;
      if (ai < aEnd) { rows_s[tid] = perm[ai]; gates_s[tid] = pgate[ai]; }
      else           { rows_s[tid] = -1;        gates_s[tid] = 0.f; }
    }
    __syncthreads();

    const __bf16* w1e = w1t + (size_t)e * H_ * D_;
    const __bf16* w2e = w2t + (size_t)e * D_ * H_;

    // prologue: kick off chunk 0 weight tiles via TDM while we gather x
    if (wave == 0) {
      tdm_load_2d((unsigned)(uintptr_t)&w1s[0][0][0], w1e, D_, HC, (unsigned long long)D_);
      tdm_load_2d((unsigned)(uintptr_t)&w2s[0][0][0], w2e, HC, D_, (unsigned long long)H_);
    }

    // gather x tile (32 rows x 512 bf16) into LDS
#pragma unroll
    for (int it = 0; it < 8; ++it) {
      int c   = it * 256 + tid;            // 16-byte chunk id
      int rm  = c >> 6;                    // 64 chunks per row
      int col = (c & 63) * 8;
      int row = rows_s[rm];
      v8bf val = {};
      if (row >= 0) val = *(const v8bf*)(xbf + (size_t)row * D_ + col);
      *(v8bf*)(&xs[rm][col]) = val;
    }

    v8f acc2[2][4];
#pragma unroll
    for (int mt = 0; mt < 2; ++mt)
#pragma unroll
      for (int nt = 0; nt < 4; ++nt)
#pragma unroll
        for (int r = 0; r < 8; ++r) acc2[mt][nt][r] = 0.f;

    for (int ch = 0; ch < NCH; ++ch) {
      const int buf = ch & 1;
      if (wave == 0) {
        const int pend = (ch + 1 < NCH);
        if (pend) {                        // prefetch next chunk into other buffer
          int hcn = (ch + 1) * HC;
          tdm_load_2d((unsigned)(uintptr_t)&w1s[buf ^ 1][0][0],
                      w1e + (size_t)hcn * D_, D_, HC, (unsigned long long)D_);
          tdm_load_2d((unsigned)(uintptr_t)&w2s[buf ^ 1][0][0],
                      w2e + hcn, HC, D_, (unsigned long long)H_);
        }
        tdm_wait(pend);                    // oldest 2 (current chunk) complete
      }
      __syncthreads();                     // weights[buf] + xs (ch==0) visible

      // ---- GEMM1: hid[32,64] = xs[32,512] @ W1[512,64] + b1, relu ----
      const int mt1 = wave & 1;
      const int nt1 = wave >> 1;
      v8f c1;
      {
        float bv = b1[e * H_ + ch * HC + nt1 * 16 + lrow];
#pragma unroll
        for (int r = 0; r < 8; ++r) c1[r] = bv;
      }
#pragma unroll
      for (int kd = 0; kd < D_ / 32; ++kd) {
        int k0 = kd * 32 + ksel;
        v16bf a = ld_op(&xs[mt1 * 16 + lrow][k0]);
        v16bf b = ld_op(&w1s[buf][nt1 * 16 + lrow][k0]);
        c1 = __builtin_amdgcn_wmma_f32_16x16x32_bf16(false, a, false, b,
                                                     (short)0, c1, false, false);
      }
      {
        int n = nt1 * 16 + lrow;
        int mbase = mt1 * 16 + (lane >> 4) * 8;
#pragma unroll
        for (int r = 0; r < 8; ++r) hs[mbase + r][n] = f2bf(fmaxf(c1[r], 0.f));
      }
      __syncthreads();                     // hs ready

      // ---- GEMM2: out[32,512] += hid[32,64] @ W2[64,512]; wave owns 64 d cols
#pragma unroll
      for (int ks = 0; ks < HC / 32; ++ks) {
        int k0 = ks * 32 + ksel;
        v16bf a0 = ld_op(&hs[lrow][k0]);
        v16bf a1 = ld_op(&hs[16 + lrow][k0]);
#pragma unroll
        for (int nt = 0; nt < 4; ++nt) {
          v16bf b = ld_op(&w2s[buf][wave * 64 + nt * 16 + lrow][k0]);
          acc2[0][nt] = __builtin_amdgcn_wmma_f32_16x16x32_bf16(
              false, a0, false, b, (short)0, acc2[0][nt], false, false);
          acc2[1][nt] = __builtin_amdgcn_wmma_f32_16x16x32_bf16(
              false, a1, false, b, (short)0, acc2[1][nt], false, false);
        }
      }
      __syncthreads();                     // all reads of buf / hs done
    }

    // gate-weighted scatter-add into output (+ b2 bias)
    const int hi = lane >> 4;
#pragma unroll
    for (int mt = 0; mt < 2; ++mt) {
#pragma unroll
      for (int nt = 0; nt < 4; ++nt) {
        int dcol = wave * 64 + nt * 16 + lrow;
        float bv = b2[e * D_ + dcol];
#pragma unroll
        for (int r = 0; r < 8; ++r) {
          int m = mt * 16 + hi * 8 + r;
          int row = rows_s[m];
          if (row >= 0) {
            float val = (acc2[mt][nt][r] + bv) * gates_s[m];
            __hip_atomic_fetch_add(out + (size_t)row * D_ + dcol, val,
                                   __ATOMIC_RELAXED, __HIP_MEMORY_SCOPE_AGENT);
          }
        }
      }
    }
  }
}

// ---------------- host launch ----------------
extern "C" void kernel_launch(void* const* d_in, const int* in_sizes, int n_in,
                              void* d_out, int out_size, void* d_ws, size_t ws_size,
                              hipStream_t stream) {
  (void)in_sizes; (void)n_in; (void)out_size; (void)ws_size;
  const float* x     = (const float*)d_in[0];
  const float* noise = (const float*)d_in[1];
  const float* Wr    = (const float*)d_in[2];
  const float* br    = (const float*)d_in[3];
  const float* Wn    = (const float*)d_in[4];
  const float* bn    = (const float*)d_in[5];
  const float* W1    = (const float*)d_in[6];
  const float* b1    = (const float*)d_in[7];
  const float* W2    = (const float*)d_in[8];
  const float* b2    = (const float*)d_in[9];

  float* out  = (float*)d_out;
  int*   topi = (int*)d_out + (size_t)LT_ * D_;

  char* ws = (char*)d_ws;
  __bf16* xbf  = (__bf16*)(ws + XBF_OFF);
  __bf16* w1t  = (__bf16*)(ws + W1T_OFF);
  __bf16* w2t  = (__bf16*)(ws + W2T_OFF);
  int*    tokExp  = (int*)(ws + TOKE_OFF);
  float*  tokGate = (float*)(ws + TOKG_OFF);
  int*    perm    = (int*)(ws + PERM_OFF);
  float*  pgate   = (float*)(ws + PGAT_OFF);
  int*    meta        = (int*)(ws + META_OFF);
  int*    counts      = meta;
  int*    cursor      = meta + 8;
  int*    offsets     = meta + 16;   // 9
  int*    tilesPrefix = meta + 25;   // 9
  int*    totalTiles  = meta + 34;   // 1

  zero_f32<<<2048, 256, 0, stream>>>(out, LT_ * D_);
  zero_i32<<<1, 64, 0, stream>>>(meta, 16);                 // counts + cursor
  cvt_x <<<2048, 256, 0, stream>>>(x, xbf, LT_ * D_);
  cvt_w1<<<2048, 256, 0, stream>>>(W1, w1t);
  cvt_w2<<<2048, 256, 0, stream>>>(W2, w2t);
  router_kernel<<<LT_ / 256, 256, 0, stream>>>(x, noise, Wr, br, Wn, bn,
                                               topi, tokExp, tokGate, counts);
  prefix_kernel<<<1, 1, 0, stream>>>(counts, offsets, tilesPrefix, totalTiles);
  scatter_kernel<<<LT_ / 256, 256, 0, stream>>>(tokExp, tokGate, offsets,
                                                cursor, perm, pgate);
  moe_gemm_kernel<<<NBLK_GEMM, 256, 0, stream>>>(xbf, w1t, w2t, b1, b2, perm,
                                                 pgate, offsets, tilesPrefix,
                                                 totalTiles, out);
}
